// DenseGATv2Layer_51342039056803
// MI455X (gfx1250) — matrix operand order repaired and checked by reading, not compiled, not run on hardware
//
#include <hip/hip_runtime.h>
#include <stdint.h>

// DenseGATv2 (decomposed GAT scores) for MI455X / gfx1250, wave32 + WMMA.
// N=4096, IN=128, HEADS=4, D=64, C=H*D=256.

typedef __attribute__((ext_vector_type(16))) _Float16 v16h;
typedef __attribute__((ext_vector_type(8)))  _Float16 v8h;
typedef __attribute__((ext_vector_type(8)))  float    v8f;

#define GAT_N     4096
#define GAT_IN    128
#define GAT_HEADS 4
#define GAT_D     64
#define GAT_C     256           // HEADS*D
#define NEG_INF_V (-9.0e15f)
#define JSPLIT    8             // waves per (head, i-block); j-dim partition

// ---------------------------------------------------------------------------
// Kernel 0: convert x -> f16, transpose W (128x256 f32) -> Wt (256x128 f16)
// ---------------------------------------------------------------------------
__global__ void gat_prep(const float* __restrict__ x, const float* __restrict__ W,
                         _Float16* __restrict__ xh, _Float16* __restrict__ WtH) {
  int tid    = blockIdx.x * blockDim.x + threadIdx.x;
  int stride = gridDim.x * blockDim.x;
  for (int i = tid; i < GAT_N * GAT_IN; i += stride) xh[i] = (_Float16)x[i];
  for (int i = tid; i < GAT_IN * GAT_C; i += stride) {
    int k = i >> 8;          // row of W (0..127)
    int n = i & 255;         // col of W (0..255)
    WtH[n * GAT_IN + k] = (_Float16)W[i];
  }
}

// ---------------------------------------------------------------------------
// Kernel 1: h = x @ W  (4096x128 @ 128x256), one wave per 16x16 output tile.
// Emits v_wmma_f32_16x16x32_f16, K-loop of 4.
// Writes h in f32 (for score dots) and h^T in f16 (B operand of aggregation).
// ---------------------------------------------------------------------------
__global__ void gat_proj(const _Float16* __restrict__ xh, const _Float16* __restrict__ WtH,
                         float* __restrict__ hf32, _Float16* __restrict__ h16T) {
  int wid  = (blockIdx.x * blockDim.x + threadIdx.x) >> 5;  // 0..4095
  int lane = threadIdx.x & 31;
  int col  = lane & 15;         // A row index == B column index
  int khi  = lane >> 4;         // which K half this lane holds
  int mb   = (wid >> 4) * 16;
  int nb   = (wid & 15) * 16;

  v8f acc = {};
  const _Float16* arow = xh  + (size_t)(mb + col) * GAT_IN;
  const _Float16* brow = WtH + (size_t)(nb + col) * GAT_IN;
#pragma unroll
  for (int ks = 0; ks < 4; ++ks) {
    int kb = ks * 32;
    // A 16x32 f16 fragment: lane holds row (lane&15); K = khi*8+e / 16+khi*8+e
    v8h a0 = *(const v8h*)(arow + kb + khi * 8);
    v8h a1 = *(const v8h*)(arow + kb + 16 + khi * 8);
    v16h a = __builtin_shufflevector(a0, a1, 0,1,2,3,4,5,6,7,8,9,10,11,12,13,14,15);
    // B 32x16 f16 fragment: lane holds column (lane&15); K = khi*8+e / 16+khi*8+e
    v8h b0h = *(const v8h*)(brow + kb + khi * 8);
    v8h b1h = *(const v8h*)(brow + kb + 16 + khi * 8);
    v16h b = __builtin_shufflevector(b0h, b1h, 0,1,2,3,4,5,6,7,8,9,10,11,12,13,14,15);
    acc = __builtin_amdgcn_wmma_f32_16x16x32_f16(false, a, false, b, (short)0,
                                                 acc, false, false);
  }
#pragma unroll
  for (int r = 0; r < 8; ++r) {
    int m = mb + khi * 8 + r;   // C row = 8*khi + r
    int c = nb + col;           // C col = lane & 15
    hf32[(size_t)m * GAT_C + c] = acc[r];
    h16T[(size_t)c * GAT_N + m] = (_Float16)acc[r];
  }
}

// ---------------------------------------------------------------------------
// Kernel 2: per-(node,head) score contributions.
//   sDstT[h*N+i] = h[i,h,:].a_src   (dest-node term "s_i")
//   sSrcT[h*N+j] = h[j,h,:].a_dst   (source-node term "s_j")
// ---------------------------------------------------------------------------
__global__ void gat_scores(const float* __restrict__ hf32,
                           const float* __restrict__ a_src,
                           const float* __restrict__ a_dst,
                           float* __restrict__ sDstT, float* __restrict__ sSrcT) {
  int tid  = blockIdx.x * blockDim.x + threadIdx.x;  // 0..16383
  int head = tid >> 12;
  int node = tid & (GAT_N - 1);
  const float4* hp = (const float4*)(hf32 + (size_t)node * GAT_C + head * GAT_D);
  const float4* as = (const float4*)a_src;
  const float4* ad = (const float4*)a_dst;
  float si = 0.f, sj = 0.f;
#pragma unroll
  for (int q = 0; q < 16; ++q) {
    float4 hv = hp[q], av = as[q], dv = ad[q];
    si += hv.x * av.x + hv.y * av.y + hv.z * av.z + hv.w * av.w;
    sj += hv.x * dv.x + hv.y * dv.y + hv.z * dv.z + hv.w * dv.w;
  }
  sDstT[tid] = si;
  sSrcT[tid] = sj;
}

// ---------------------------------------------------------------------------
// Kernel 3: flash-style masked softmax + aggregation.
// One 8-wave block per (head, 16-row dest block); each wave owns a strided
// j-partition (online softmax + WMMA accumulation), partials merged in LDS:
//   M = max_w m_w ;  out = sum_w e^{m_w-M} acc_w / sum_w e^{m_w-M} l_w
// ---------------------------------------------------------------------------
__global__ void __launch_bounds__(256)
gat_flash(const unsigned char* __restrict__ adj,   // jnp.bool_ -> 1 byte/elem
          const float* __restrict__ sDstT, const float* __restrict__ sSrcT,
          const _Float16* __restrict__ h16T, float* __restrict__ out) {
  __shared__ float s_m[JSPLIT][16];
  __shared__ float s_l[JSPLIT][16];
  __shared__ float s_acc[JSPLIT][16][GAT_D];       // 32 KB

  int bid  = blockIdx.x;                 // 0..1023: (head, iblock)
  int head = bid >> 8;
  int ib   = (bid & 255) * 16;
  int w    = threadIdx.x >> 5;           // wave in block = j-partition id
  int lane = threadIdx.x & 31;
  int row  = lane & 15;                  // P-row for this lane, also B column
  int khi  = lane >> 4;

  float si = sDstT[head * GAT_N + ib + row];                   // loop invariant
  const float*         ssrc  = sSrcT + head * GAT_N;
  const unsigned char* mrow  = adj + (size_t)(ib + row) * GAT_N;
  const _Float16*      bbase = h16T + (size_t)(head * GAT_D + row) * GAT_N;

  float m_p = -__builtin_inff();   // running max, P layout (row = lane&15)
  float l_p = 0.f;                 // running sum, P layout
  float m_c[8];                    // running max, C layout (rows khi*8+r)
#pragma unroll
  for (int r = 0; r < 8; ++r) m_c[r] = -__builtin_inff();
  v8f acc0 = {}, acc1 = {}, acc2 = {}, acc3 = {};

  for (int jb = w * 32; jb < GAT_N; jb += JSPLIT * 32) {
    if (jb + JSPLIT * 32 < GAT_N)
      __builtin_prefetch(mrow + jb + JSPLIT * 32, 0, 1);   // global_prefetch_b8

    // --- adjacency bytes for this lane's 16 (row, K) positions ---
    uint32_t mk[4];
    { const uint32_t* mp0 = (const uint32_t*)(mrow + jb + khi * 8);
      const uint32_t* mp1 = (const uint32_t*)(mrow + jb + 16 + khi * 8);
      mk[0] = mp0[0]; mk[1] = mp0[1]; mk[2] = mp1[0]; mk[3] = mp1[1]; }

    // --- source-node score terms at the same K positions ---
    float sj[16];
    { const float4* sp0 = (const float4*)(ssrc + jb + khi * 8);
      const float4* sp1 = (const float4*)(ssrc + jb + 16 + khi * 8);
      float4 v0 = sp0[0], v1 = sp0[1], v2 = sp1[0], v3 = sp1[1];
      sj[0]=v0.x;  sj[1]=v0.y;  sj[2]=v0.z;  sj[3]=v0.w;
      sj[4]=v1.x;  sj[5]=v1.y;  sj[6]=v1.z;  sj[7]=v1.w;
      sj[8]=v2.x;  sj[9]=v2.y;  sj[10]=v2.z; sj[11]=v2.w;
      sj[12]=v3.x; sj[13]=v3.y; sj[14]=v3.z; sj[15]=v3.w; }

    // --- scores: leaky_relu(s_i + s_j, 0.2), masked to -9e15 ---
    float sc[16];
    float lmax = -__builtin_inff();
#pragma unroll
    for (int e = 0; e < 16; ++e) {
      float v = si + sj[e];
      v = (v > 0.f) ? v : 0.2f * v;
      unsigned byte = (mk[e >> 2] >> ((e & 3) * 8)) & 0xFFu;
      v = byte ? v : NEG_INF_V;
      sc[e] = v;
      lmax = fmaxf(lmax, v);
    }
    float cmax = fmaxf(lmax, __shfl_xor(lmax, 16, 32));  // full-row chunk max
    float mnew = fmaxf(m_p, cmax);

    // --- rescale accumulators (C layout: lane owns rows khi*8+r) ---
#pragma unroll
    for (int r = 0; r < 8; ++r) {
      float mr  = __shfl(mnew, khi * 8 + r, 32);  // lane q holds row q's stats
      float sca = __expf(m_c[r] - mr);
      m_c[r] = mr;
      acc0[r] *= sca; acc1[r] *= sca; acc2[r] *= sca; acc3[r] *= sca;
    }

    // --- p = exp(score - newmax), build A fragment, row sums ---
    float ls = 0.f;
    v16h a;
#pragma unroll
    for (int e = 0; e < 16; ++e) {
      float p = __expf(sc[e] - mnew);   // fully-masked row -> exp(0)=1 uniform
      ls += p;
      a[e] = (_Float16)p;
    }
    float csum = ls + __shfl_xor(ls, 16, 32);
    l_p = l_p * __expf(m_p - mnew) + csum;
    m_p = mnew;

    // --- B fragments: h^T rows are contiguous in j -> 16B vector loads, K = khi*8+e / 16+khi*8+e ---
    const _Float16* bp = bbase + jb + khi * 8;
    v16h b0 = __builtin_shufflevector(*(const v8h*)(bp), *(const v8h*)(bp + 16), 0,1,2,3,4,5,6,7,8,9,10,11,12,13,14,15);
    v16h b1 = __builtin_shufflevector(*(const v8h*)(bp + 16 * GAT_N), *(const v8h*)(bp + 16 * GAT_N + 16), 0,1,2,3,4,5,6,7,8,9,10,11,12,13,14,15);
    v16h b2 = __builtin_shufflevector(*(const v8h*)(bp + 32 * GAT_N), *(const v8h*)(bp + 32 * GAT_N + 16), 0,1,2,3,4,5,6,7,8,9,10,11,12,13,14,15);
    v16h b3 = __builtin_shufflevector(*(const v8h*)(bp + 48 * GAT_N), *(const v8h*)(bp + 48 * GAT_N + 16), 0,1,2,3,4,5,6,7,8,9,10,11,12,13,14,15);
    acc0 = __builtin_amdgcn_wmma_f32_16x16x32_f16(false, a, false, b0, (short)0, acc0, false, false);
    acc1 = __builtin_amdgcn_wmma_f32_16x16x32_f16(false, a, false, b1, (short)0, acc1, false, false);
    acc2 = __builtin_amdgcn_wmma_f32_16x16x32_f16(false, a, false, b2, (short)0, acc2, false, false);
    acc3 = __builtin_amdgcn_wmma_f32_16x16x32_f16(false, a, false, b3, (short)0, acc3, false, false);
  }

  // --- dump this wave's partial state to LDS ---
  if (khi == 0) {               // lanes 0..15 hold row-lane stats
    s_m[w][row] = m_p;
    s_l[w][row] = l_p;
  }
#pragma unroll
  for (int r = 0; r < 8; ++r) {
    int rr = khi * 8 + r;
    s_acc[w][rr][ 0 + row] = acc0[r];
    s_acc[w][rr][16 + row] = acc1[r];
    s_acc[w][rr][32 + row] = acc2[r];
    s_acc[w][rr][48 + row] = acc3[r];
  }
  __syncthreads();

  // --- merge JSPLIT partials: thread t owns (row = t/16, cols 4*(t%16)..+3) ---
  int t    = threadIdx.x;
  int orow = t >> 4;
  int cb   = (t & 15) * 4;
  float M = -__builtin_inff();
#pragma unroll
  for (int ww = 0; ww < JSPLIT; ++ww) M = fmaxf(M, s_m[ww][orow]);
  float L = 0.f;
  float4 o = make_float4(0.f, 0.f, 0.f, 0.f);
#pragma unroll
  for (int ww = 0; ww < JSPLIT; ++ww) {
    float sca = __expf(s_m[ww][orow] - M);
    L += s_l[ww][orow] * sca;
    float4 av = *(const float4*)&s_acc[ww][orow][cb];
    o.x += sca * av.x; o.y += sca * av.y; o.z += sca * av.z; o.w += sca * av.w;
  }
  float inv = 1.0f / L;
  float* op = out + (size_t)(ib + orow) * GAT_C + head * GAT_D + cb;
  *(float4*)op = make_float4(o.x * inv, o.y * inv, o.z * inv, o.w * inv);
}

// ---------------------------------------------------------------------------
extern "C" void kernel_launch(void* const* d_in, const int* in_sizes, int n_in,
                              void* d_out, int out_size, void* d_ws, size_t ws_size,
                              hipStream_t stream) {
  const float*         x     = (const float*)d_in[0];
  const unsigned char* adj   = (const unsigned char*)d_in[1];  // jnp.bool_ = 1B
  const float*         W     = (const float*)d_in[2];
  const float*         a_src = (const float*)d_in[3];
  const float*         a_dst = (const float*)d_in[4];
  float*               out   = (float*)d_out;

  char*  ws  = (char*)d_ws;
  size_t off = 0;
  auto alloc = [&](size_t bytes) -> void* {
    void* p = ws + off;
    off = (off + bytes + 255) & ~(size_t)255;
    return p;
  };
  _Float16* xh    = (_Float16*)alloc((size_t)GAT_N * GAT_IN * 2);   // 1 MB
  _Float16* WtH   = (_Float16*)alloc((size_t)GAT_C * GAT_IN * 2);   // 64 KB
  _Float16* h16T  = (_Float16*)alloc((size_t)GAT_C * GAT_N * 2);    // 2 MB
  float*    hf32  = (float*)alloc((size_t)GAT_N * GAT_C * 4);       // 4 MB
  float*    sDstT = (float*)alloc((size_t)GAT_HEADS * GAT_N * 4);   // 64 KB
  float*    sSrcT = (float*)alloc((size_t)GAT_HEADS * GAT_N * 4);   // 64 KB

  gat_prep  <<<512, 256, 0, stream>>>(x, W, xh, WtH);
  gat_proj  <<<(4096 * 32) / 256, 256, 0, stream>>>(xh, WtH, hf32, h16T); // 4096 waves
  gat_scores<<<(GAT_HEADS * GAT_N) / 256, 256, 0, stream>>>(hf32, a_src, a_dst,
                                                            sDstT, sSrcT);
  // 1024 blocks x 8 waves = 8192 waves
  gat_flash <<<GAT_HEADS * (GAT_N / 16), 256, 0, stream>>>(adj, sDstT, sSrcT,
                                                           h16T, out);
}
